// MHAKernel_6691559047308
// MI455X (gfx1250) — matrix operand order repaired and checked by reading, not compiled, not run on hardware
//
#include <hip/hip_runtime.h>

typedef __attribute__((ext_vector_type(16))) __bf16 v16bf;
typedef __attribute__((ext_vector_type(8)))  __bf16 v8bf;
typedef __attribute__((ext_vector_type(8)))  float  v8f;

__device__ __forceinline__ v8f wmma_bf(v16bf a, v16bf b, v8f c) {
    return __builtin_amdgcn_wmma_f32_16x16x32_bf16(
        /*neg_a=*/false, a, /*neg_b=*/false, b,
        /*c_mod=*/(short)0, c, /*reuse_a=*/false, /*reuse_b=*/false);
}

// DPP16 cross-lane move (stays within each 16-lane half of the wave)
template <int CTRL>
__device__ __forceinline__ float dpp_mov(float x) {
    return __builtin_bit_cast(float,
        __builtin_amdgcn_update_dpp(0, __builtin_bit_cast(int, x),
                                    CTRL, 0xF, 0xF, true));
}
#define DPP_XOR1  0xB1   // quad_perm(1,0,3,2)
#define DPP_XOR2  0x4E   // quad_perm(2,3,0,1)
#define DPP_MIR8  0x141  // row_half_mirror (lane^7 within 8)
#define DPP_MIR16 0x140  // row_mirror      (lane^15 within 16)

__device__ __forceinline__ float redmax16(float x) {
    x = fmaxf(x, dpp_mov<DPP_XOR1>(x));
    x = fmaxf(x, dpp_mov<DPP_XOR2>(x));
    x = fmaxf(x, dpp_mov<DPP_MIR8>(x));
    x = fmaxf(x, dpp_mov<DPP_MIR16>(x));
    return x;
}
__device__ __forceinline__ float redsum16(float x) {
    x += dpp_mov<DPP_XOR1>(x);
    x += dpp_mov<DPP_XOR2>(x);
    x += dpp_mov<DPP_MIR8>(x);
    x += dpp_mov<DPP_MIR16>(x);
    return x;
}

// 1/sqrt(128) * log2(e): scores live in log2 domain -> bare v_exp_f32
#define SM_SCALE_LOG2E 0.1275187928509809f

__global__ __launch_bounds__(128)
void fa_fwd_kernel(const float* __restrict__ Q, const float* __restrict__ K,
                   const float* __restrict__ V, float* __restrict__ O,
                   int B, int S, int H) {
    constexpr int BM = 64, BN = 64, DH = 128;
    const int nQB = S / BM;
    const int qb = blockIdx.x % nQB;
    const int h  = (blockIdx.x / nQB) % H;
    const int b  = blockIdx.x / (nQB * H);
    const int q0 = qb * BM;

    const int tid   = threadIdx.x;
    const int wv    = tid >> 5;
    const int lane  = tid & 31;
    const int lhalf = lane >> 4;          // 0 or 1
    const int l15   = lane & 15;

    __shared__ __align__(32) __bf16 Klds[BN][DH];     // 16 KB, keys row-major
    __shared__ __align__(32) __bf16 VT[DH][BN];       // 16 KB, V transposed
    __shared__ __align__(32) __bf16 Plds[4][16][BN];  //  8 KB, per-wave P scratch

    // ---- Q tile for this wave, directly in WMMA 16-bit A-operand layout ----
    v16bf qf[4];
    {
        const int m = q0 + (wv << 4) + l15;
        const float* qp = Q + (((size_t)b * S + m) * H + h) * DH;
        const int baseK = lhalf * 16;
#pragma unroll
        for (int f = 0; f < 4; ++f) {
            const int d0 = f * 32 + baseK;
#pragma unroll
            for (int i = 0; i < 8; ++i) qf[f][i]     = (__bf16)(qp[d0 + i]      * SM_SCALE_LOG2E);
#pragma unroll
            for (int i = 0; i < 8; ++i) qf[f][8 + i] = (__bf16)(qp[d0 + 8 + i] * SM_SCALE_LOG2E);
        }
    }

    // ---- online-softmax state (log2 domain): rows m = 8*lhalf + r ----
    float mrow[8], lrow[8];
    v8f acc[8] = {};
#pragma unroll
    for (int r = 0; r < 8; ++r) { mrow[r] = -1e30f; lrow[r] = 0.0f; }

    const int rowMin  = q0 + (wv << 4);
    const int rowLast = rowMin + 15;
    const int nkb = q0 / BN + 1;

    for (int kb = 0; kb < nkb; ++kb) {
        __syncthreads();   // previous tile fully consumed before restaging
        // ---- cooperative staging ----
        {
            // K: thread covers half a key row (64 dims), contiguous loads + stores
            const int row  = tid >> 1;
            const int dseg = (tid & 1) * 64;
            const float* kp = K + (((size_t)b * S + kb * BN + row) * H + h) * DH + dseg;
            if (kb + 1 < nkb) __builtin_prefetch(kp + (size_t)BN * H * DH, 0, 0);
#pragma unroll
            for (int j = 0; j < 64; ++j) Klds[row][dseg + j] = (__bf16)kp[j];

            // V: thread owns dim d = tid; row reads coalesced across block,
            // transposed store is contiguous 128B -> ds_store_b128s
            const float* vp = V + ((size_t)b * S + kb * BN) * (size_t)(H * DH) + h * DH + tid;
            if (kb + 1 < nkb) __builtin_prefetch(vp + (size_t)BN * H * DH, 0, 0);
#pragma unroll
            for (int j = 0; j < 64; ++j) VT[tid][j] = (__bf16)vp[(size_t)j * H * DH];
        }
        __syncthreads();

        if (kb * BN <= rowLast) {
            // ---- S = Q K^T : four 16x16 score tiles over 64 keys ----
            // software-pipelined B-fragment loads: issue frag i+1 before wmma i
            v8f s[4] = {};
            {
                auto kfrag = [&](int i) -> v16bf {
                    const int f = i >> 2, nt = i & 3;
                    return *(const v16bf*)&Klds[nt * 16 + l15][f * 32 + lhalf * 16];
                };
                v16bf cur = kfrag(0), nxt;
#pragma unroll
                for (int i = 0; i < 16; ++i) {
                    if (i < 15) nxt = kfrag(i + 1);
                    s[i & 3] = wmma_bf(qf[i >> 2], cur, s[i & 3]);
                    cur = nxt;
                }
            }

            // ---- causal mask (diagonal tiles only; wave-uniform branch) ----
            const int rbase = rowMin + (lhalf << 3);
            if (kb * BN + (BN - 1) > rowMin) {
#pragma unroll
                for (int nt = 0; nt < 4; ++nt) {
                    const int keyg = kb * BN + nt * 16 + l15;
#pragma unroll
                    for (int r = 0; r < 8; ++r)
                        s[nt][r] = (keyg <= rbase + r) ? s[nt][r] : -1e30f;
                }
            }

            // ---- online softmax (log2 domain, DPP butterflies) ----
            float rsum[8];
            v8f sv;   // per-row rescale factors, applied as packed vector muls
#pragma unroll
            for (int r = 0; r < 8; ++r) {
                const float t = redmax16(
                    fmaxf(fmaxf(s[0][r], s[1][r]), fmaxf(s[2][r], s[3][r])));
                const float mnew  = fmaxf(mrow[r], t);
                const float scale = __builtin_amdgcn_exp2f(mrow[r] - mnew);
                mrow[r] = mnew;
                sv[r] = scale;
                float ps = 0.0f;
#pragma unroll
                for (int nt = 0; nt < 4; ++nt) {
                    const float p = __builtin_amdgcn_exp2f(s[nt][r] - mnew);
                    s[nt][r] = p;
                    ps += p;
                }
                rsum[r] = ps;
                lrow[r] *= scale;
            }
#pragma unroll
            for (int t2 = 0; t2 < 8; ++t2) acc[t2] *= sv;   // v_pk_mul_f32
#pragma unroll
            for (int r = 0; r < 8; ++r)
                lrow[r] += redsum16(rsum[r]);

            // ---- reshape P: C-layout -> A-layout via per-wave LDS scratch ----
#pragma unroll
            for (int r = 0; r < 8; ++r) {
                const int m = (lhalf << 3) + r;
#pragma unroll
                for (int nt = 0; nt < 4; ++nt)
                    Plds[wv][m][nt * 16 + l15] = (__bf16)s[nt][r];
            }
            asm volatile("s_wait_dscnt 0" ::: "memory");   // intra-wave LDS RAW

            const __bf16* pp = &Plds[wv][l15][0];
            const int baseK = lhalf * 16;
            v16bf pf[2];
#pragma unroll
            for (int c = 0; c < 2; ++c) {
                v8bf plo = *(const v8bf*)(pp + c * 32 + baseK);
                v8bf phi = *(const v8bf*)(pp + c * 32 + baseK + 8);
#pragma unroll
                for (int i = 0; i < 8; ++i) { pf[c][i] = plo[i]; pf[c][8 + i] = phi[i]; }
            }

            // ---- O += P V : 8 d-tiles x 2 k-chunks, pipelined like QK ----
            {
                auto vfrag = [&](int i) -> v16bf {
                    const int t = i >> 1, c = i & 1;
                    return *(const v16bf*)&VT[(t << 4) + l15][c * 32 + lhalf * 16];
                };
                v16bf cur = vfrag(0), nxt;
#pragma unroll
                for (int i = 0; i < 16; ++i) {
                    if (i < 15) nxt = vfrag(i + 1);
                    acc[i >> 1] = wmma_bf(pf[i & 1], cur, acc[i >> 1]);
                    cur = nxt;
                }
            }
        }
    }

    // ---- epilogue: normalize (packed) and store fp32 [B,S,H,D] ----
    v8f iv;
#pragma unroll
    for (int r = 0; r < 8; ++r) iv[r] = __builtin_amdgcn_rcpf(lrow[r]);
#pragma unroll
    for (int t = 0; t < 8; ++t) acc[t] *= iv;   // v_pk_mul_f32

    const int mbase = q0 + (wv << 4) + (lhalf << 3);
#pragma unroll
    for (int r = 0; r < 8; ++r) {
        float* op = O + (((size_t)b * S + (mbase + r)) * H + h) * 128 + l15;
#pragma unroll
        for (int t = 0; t < 8; ++t)
            op[t * 16] = acc[t][r];
    }
}

extern "C" void kernel_launch(void* const* d_in, const int* in_sizes, int n_in,
                              void* d_out, int out_size, void* d_ws, size_t ws_size,
                              hipStream_t stream) {
    (void)in_sizes; (void)n_in; (void)out_size; (void)d_ws; (void)ws_size;
    const float* q = (const float*)d_in[0];
    const float* k = (const float*)d_in[1];
    const float* v = (const float*)d_in[2];
    float* out = (float*)d_out;
    const int B = 4, S = 2048, H = 16;
    dim3 grid(B * H * (S / 64)), block(128);
    hipLaunchKernelGGL(fa_fwd_kernel, grid, block, 0, stream, q, k, v, out, B, S, H);
}
